// CNSDFM_62088047231686
// MI455X (gfx1250) — matrix operand order repaired and checked by reading, not compiled, not run on hardware
//
#include <hip/hip_runtime.h>
#include <hip/hip_bf16.h>
#include <cstdint>

// Problem dims
#define Bb 64
#define Cc 256
#define Tt 512
#define Hh 128
#define Gg 512   // 4*H

typedef __attribute__((ext_vector_type(16))) _Float16 v16h;
typedef __attribute__((ext_vector_type(8)))  _Float16 v8h;
typedef __attribute__((ext_vector_type(8)))  float    v8f;

// ---------------------------------------------------------------------------
// WMMA fragment helpers (contiguous-k A/B order, wave32, 16x16x32 f16 -> f32)
// A (16x32): lane row m = m0 + lane%16; one v16h at k0 + 16*(lane/16).
// B (32x16): lane col n = n0 + lane%16; one v16h at k0 + 16*(lane/16).
// C/D (16x16 f32): acc[r] <-> row m0 + r + 8*(lane/16), col n0 + lane%16.
// Fragment-native f32 tensor layout: [...tile...][lane][r] (v8f per lane).
// ---------------------------------------------------------------------------
__device__ __forceinline__ v16h load_a_frag(const _Float16* A, int lda,
                                            int mrow, int k0, int half) {
  const _Float16* p = A + (size_t)mrow * lda + k0 + 16 * half;
  return *(const v16h*)p;
}

__device__ __forceinline__ v16h load_b_frag(const _Float16* W, int ldw,
                                            int nrow, int k0, int half) {
  return *(const v16h*)(W + (size_t)nrow * ldw + k0 + 16 * half);
}

__device__ __forceinline__ v8f wmma_f16(v16h a, v16h b, v8f c) {
  return __builtin_amdgcn_wmma_f32_16x16x32_f16(false, a, false, b,
                                                (short)0, c, false, false);
}

__device__ __forceinline__ float sigm(float x) { return 1.f / (1.f + __expf(-x)); }
__device__ __forceinline__ float softp(float x) {
  return x > 20.f ? x : log1pf(__expf(x));
}

// Accumulate one 16x64 tile of  A[M,K] @ W[N,K]^T + bias  into acc[4].
__device__ __forceinline__ void gemm_tile16x64(const _Float16* A, int lda,
                                               const _Float16* W, int ldw,
                                               const float* bias,
                                               int m0, int n0, int K,
                                               int lane, v8f acc[4]) {
  int half = lane >> 4, l16 = lane & 15;
  int mrow = m0 + l16;
#pragma unroll
  for (int j = 0; j < 4; ++j) {
    float bv = bias[n0 + j * 16 + l16];
#pragma unroll
    for (int r = 0; r < 8; ++r) acc[j][r] = bv;
  }
  for (int k0 = 0; k0 < K; k0 += 32) {
    v16h a = load_a_frag(A, lda, mrow, k0, half);
#pragma unroll
    for (int j = 0; j < 4; ++j) {
      v16h b = load_b_frag(W, ldw, n0 + j * 16 + l16, k0, half);
      acc[j] = wmma_f16(a, b, acc[j]);
    }
  }
}

// ---------------------------------------------------------------------------
// Prep kernels
// ---------------------------------------------------------------------------
__global__ void k_cvt(const float* __restrict__ s, _Float16* __restrict__ d, int n) {
  int i = blockIdx.x * blockDim.x + threadIdx.x;
  if (i < n) d[i] = (_Float16)s[i];
}

__global__ void k_badd(const float* __restrict__ a, const float* __restrict__ b,
                       float* __restrict__ o) {
  int i = blockIdx.x * blockDim.x + threadIdx.x;
  if (i < Gg) o[i] = a[i] + b[i];
}

// zT16[t][b][c] = (f16) z[b][c][t]   (row-major A for the gatesX GEMM)
__global__ void k_zt(const float* __restrict__ z, _Float16* __restrict__ zt16) {
  int i = blockIdx.x * blockDim.x + threadIdx.x;
  if (i >= Tt * Bb * Cc) return;
  int c = i & (Cc - 1);
  int b = (i >> 8) & (Bb - 1);
  int t = i >> 14;
  zt16[i] = (_Float16)z[((size_t)b * Cc + c) * Tt + t];
}

// zf: fragment-layout f32 copy of z_tbc for the filter scan.
// o = ((((t*4 + mt)*16 + ct)*32 + lane)*8 + r)
__global__ void k_zf(const float* __restrict__ z, float* __restrict__ zf) {
  int o = blockIdx.x * blockDim.x + threadIdx.x;
  if (o >= Tt * Bb * Cc) return;
  int r = o & 7;
  int lane = (o >> 3) & 31;
  int ct = (o >> 8) & 15;
  int mt = (o >> 12) & 3;
  int t = o >> 14;
  int m = mt * 16 + (lane >> 4) * 8 + r;
  int c = ct * 16 + (lane & 15);
  zf[o] = z[((size_t)m * Cc + c) * Tt + t];
}

// ---------------------------------------------------------------------------
// Phase 1: gatesX = zT @ Wih^T + b_all  (M=T*B, N=512, K=256)
// Output in fragment layout gxf[((t*4+mt)*32 + ntile)*32 + lane][8].
// ---------------------------------------------------------------------------
__global__ void __launch_bounds__(256)
k_gatesx(const _Float16* __restrict__ zt16, const _Float16* __restrict__ wih16,
         const float* __restrict__ ball, float* __restrict__ gxf) {
  int lane = threadIdx.x & 31, w = threadIdx.x >> 5;
  int g = blockIdx.x * 8 + w;           // 16384 tiles of 16x64
  int m0 = (g >> 3) * 16;
  int n0 = (g & 7) * 64;
  v8f acc[4];
  gemm_tile16x64(zt16, Cc, wih16, Cc, ball, m0, n0, Cc, lane, acc);
  int t = m0 >> 6, mt = (m0 >> 4) & 3;
#pragma unroll
  for (int j = 0; j < 4; ++j) {
    int ntile = (g & 7) * 4 + j;
    size_t idx = ((((size_t)t * 4 + mt) * 32 + ntile) * 32 + lane) * 8;
    *(v8f*)(gxf + idx) = acc[j];
  }
}

// ---------------------------------------------------------------------------
// Phase 2: persistent LSTM recurrence, 1 block x 512 threads (16 waves).
// Wave w: jt = w>>1 (hidden col tile), mts = {0,1} or {2,3} by (w&1).
// Whh B-fragments (16 x v16h) pinned in VGPRs for all 512 steps.
// Cell state in VGPRs; h double-buffered in LDS.
// ---------------------------------------------------------------------------
__global__ void __launch_bounds__(512)
k_lstm(const float* __restrict__ gxf, const _Float16* __restrict__ whh16,
       _Float16* __restrict__ hall16) {
  __shared__ __align__(32) _Float16 hbuf[2][Bb][Hh];   // 32 KB
  int lane = threadIdx.x & 31, w = threadIdx.x >> 5;
  int half = lane >> 4, l16 = lane & 15;
  int jt = w >> 1;
  int mtb = (w & 1) * 2;

  for (int i = threadIdx.x; i < 2 * Bb * Hh; i += 512)
    (&hbuf[0][0][0])[i] = (_Float16)0.f;

  // preload recurrent-weight B fragments: bw[q][ki]
  v16h bw[4][4];
#pragma unroll
  for (int q = 0; q < 4; ++q)
#pragma unroll
    for (int ki = 0; ki < 4; ++ki)
      bw[q][ki] = load_b_frag(whh16, Hh, q * 128 + jt * 16 + l16, ki * 32, half);

  v8f cfrag[2];
#pragma unroll
  for (int u = 0; u < 2; ++u)
#pragma unroll
    for (int r = 0; r < 8; ++r) cfrag[u][r] = 0.f;
  __syncthreads();

  for (int t = 0; t < Tt; ++t) {
    int cur = t & 1, nxt = cur ^ 1;
#pragma unroll
    for (int im = 0; im < 2; ++im) {
      int mt = mtb + im;
      int m0 = mt * 16;
      int mrow = m0 + l16;
      v8f acc[4];
#pragma unroll
      for (int q = 0; q < 4; ++q) {
        size_t idx = ((((size_t)t * 4 + mt) * 32 + (q * 8 + jt)) * 32 + lane) * 8;
        acc[q] = *(const v8f*)(gxf + idx);
      }
#pragma unroll
      for (int ki = 0; ki < 4; ++ki) {
        v16h a = load_a_frag(&hbuf[cur][0][0], Hh, mrow, ki * 32, half);
#pragma unroll
        for (int q = 0; q < 4; ++q)
          acc[q] = wmma_f16(a, bw[q][ki], acc[q]);
      }
#pragma unroll
      for (int r = 0; r < 8; ++r) {
        float ig = sigm(acc[0][r]);
        float fg = sigm(acc[1][r]);
        float gg = tanhf(acc[2][r]);
        float og = sigm(acc[3][r]);
        float cv = fg * cfrag[im][r] + ig * gg;
        cfrag[im][r] = cv;
        float hv = og * tanhf(cv);
        hbuf[nxt][m0 + r + 8 * half][jt * 16 + l16] = (_Float16)hv;
      }
    }
    __syncthreads();
    // coalesced copy h(t) -> global (reads only; next step also only reads nxt)
    {
      const v8h* src = (const v8h*)&hbuf[nxt][0][0];
      v8h* dst = (v8h*)(hall16 + (size_t)t * Bb * Hh);
      for (int i = threadIdx.x; i < (Bb * Hh) / 8; i += 512) dst[i] = src[i];
    }
  }
}

// ---------------------------------------------------------------------------
// Phase 3: resid16[m][n] = z_t[b][n] - (h_all[m] @ fc_W^T + fc_b)
// ---------------------------------------------------------------------------
__global__ void __launch_bounds__(256)
k_fc_resid(const _Float16* __restrict__ hall16, const _Float16* __restrict__ fcw16,
           const float* __restrict__ fcb, const float* __restrict__ z,
           _Float16* __restrict__ res16) {
  int lane = threadIdx.x & 31, w = threadIdx.x >> 5;
  int half = lane >> 4, l16 = lane & 15;
  int g = blockIdx.x * 8 + w;           // 8192 tiles
  int m0 = (g >> 2) * 16;
  int n0 = (g & 3) * 64;
  v8f acc[4];
  gemm_tile16x64(hall16, Hh, fcw16, Hh, fcb, m0, n0, Hh, lane, acc);
#pragma unroll
  for (int j = 0; j < 4; ++j) {
    int n = n0 + j * 16 + l16;
#pragma unroll
    for (int r = 0; r < 8; ++r) {
      int m = m0 + r + 8 * half;
      int t = m >> 6, b = m & (Bb - 1);
      float zv = z[((size_t)b * Cc + n) * Tt + t];
      res16[(size_t)m * Cc + n] = (_Float16)(zv - acc[j][r]);
    }
  }
}

// Phase 4a: nh1 = relu(resid @ nW1^T + nb1)
__global__ void __launch_bounds__(256)
k_noise1(const _Float16* __restrict__ res16, const _Float16* __restrict__ nw116,
         const float* __restrict__ nb1, _Float16* __restrict__ nh116) {
  int lane = threadIdx.x & 31, w = threadIdx.x >> 5;
  int half = lane >> 4, l16 = lane & 15;
  int g = blockIdx.x * 8 + w;
  int m0 = (g >> 2) * 16;
  int n0 = (g & 3) * 64;
  v8f acc[4];
  gemm_tile16x64(res16, Cc, nw116, Cc, nb1, m0, n0, Cc, lane, acc);
#pragma unroll
  for (int j = 0; j < 4; ++j) {
    int n = n0 + j * 16 + l16;
#pragma unroll
    for (int r = 0; r < 8; ++r)
      nh116[(size_t)(m0 + r + 8 * half) * Cc + n] = (_Float16)fmaxf(acc[j][r], 0.f);
  }
}

// Phase 4b: K = sigmoid(nh1 @ nW2^T + nb2)
// K stored fragment-layout kf[((t*4+mt)*16 + ntile)*32 + lane][8];
// also writes uncert[b][n][t] = 1-K for t >= 1.
__global__ void __launch_bounds__(256)
k_noise2(const _Float16* __restrict__ nh116, const _Float16* __restrict__ nw216,
         const float* __restrict__ nb2, float* __restrict__ kf,
         float* __restrict__ uncert) {
  int lane = threadIdx.x & 31, w = threadIdx.x >> 5;
  int half = lane >> 4, l16 = lane & 15;
  int g = blockIdx.x * 8 + w;
  int m0 = (g >> 2) * 16;
  int n0 = (g & 3) * 64;
  v8f acc[4];
  gemm_tile16x64(nh116, Cc, nw216, Cc, nb2, m0, n0, Cc, lane, acc);
  int t = m0 >> 6, mt = (m0 >> 4) & 3;
#pragma unroll
  for (int j = 0; j < 4; ++j) {
    int n = n0 + j * 16 + l16;
    int ntile = (g & 3) * 4 + j;
    v8f kv8;
#pragma unroll
    for (int r = 0; r < 8; ++r) kv8[r] = sigm(acc[j][r]);
    size_t idx = ((((size_t)t * 4 + mt) * 16 + ntile) * 32 + lane) * 8;
    *(v8f*)(kf + idx) = kv8;
    if (t >= 1) {
#pragma unroll
      for (int r = 0; r < 8; ++r) {
        int b = (m0 + r + 8 * half) & (Bb - 1);
        uncert[((size_t)b * Cc + n) * Tt + t] = 1.f - kv8[r];
      }
    }
  }
}

// ---------------------------------------------------------------------------
// Phase 5: persistent filter scan, 1 block x 512 threads (16 waves).
// Wave w owns column tile ct = w (16 cols); x state (4 mt x v8f) in VGPRs.
// Stage-1 weights: wave w handles mat = w>>3, nt2 = w&7 (B-frags in VGPRs).
// Stage-2 weights: per-wave column slice of dW2/fW2 (B-frags in VGPRs).
// ---------------------------------------------------------------------------
__global__ void __launch_bounds__(512)
k_filter(const float* __restrict__ zf, const float* __restrict__ nz,
         const float* __restrict__ kf,
         const _Float16* __restrict__ dw1, const float* __restrict__ db1,
         const _Float16* __restrict__ dw2, const float* __restrict__ db2,
         const _Float16* __restrict__ fw1, const float* __restrict__ fb1,
         const _Float16* __restrict__ fw2, const float* __restrict__ fb2,
         float* __restrict__ refined, float* __restrict__ uncert) {
  __shared__ __align__(32) _Float16 xbuf[Bb][Cc];   // 32 KB
  __shared__ __align__(32) _Float16 hd[Bb][Hh];     // 16 KB
  __shared__ __align__(32) _Float16 hf[Bb][Hh];     // 16 KB
  int lane = threadIdx.x & 31, w = threadIdx.x >> 5;
  int half = lane >> 4, l16 = lane & 15;
  int ct = w;                   // column tile (0..15) for x-state & stage 2
  int c = ct * 16 + l16;
  int mat = w >> 3, nt2 = w & 7;  // stage-1 tile ownership
  int nh = nt2 * 16 + l16;

  // persistent B fragments
  const _Float16* W1 = mat ? fw1 : dw1;
  float b1v = (mat ? fb1 : db1)[nh];
  v16h bw1[8];
#pragma unroll
  for (int ki = 0; ki < 8; ++ki)
    bw1[ki] = load_b_frag(W1, Cc, nh, ki * 32, half);
  v16h bwd2[4], bwf2[4];
#pragma unroll
  for (int ki = 0; ki < 4; ++ki) {
    bwd2[ki] = load_b_frag(dw2, Hh, c, ki * 32, half);
    bwf2[ki] = load_b_frag(fw2, Hh, c, ki * 32, half);
  }
  float db2v = db2[c], fb2v = fb2[c];

  // x0 = z[:, :, 0]; write t=0 output slices
  v8f xr[4];
#pragma unroll
  for (int mt = 0; mt < 4; ++mt) {
    size_t idx = ((((size_t)0 * 4 + mt) * 16 + ct) * 32 + lane) * 8;
    xr[mt] = *(const v8f*)(zf + idx);
#pragma unroll
    for (int r = 0; r < 8; ++r) {
      int m = mt * 16 + r + 8 * half;
      refined[((size_t)m * Cc + c) * Tt + 0] = xr[mt][r];
      uncert [((size_t)m * Cc + c) * Tt + 0] = 0.f;
    }
  }

  for (int t = 1; t < Tt; ++t) {
    // (a) publish x as f16
#pragma unroll
    for (int mt = 0; mt < 4; ++mt)
#pragma unroll
      for (int r = 0; r < 8; ++r)
        xbuf[mt * 16 + r + 8 * half][c] = (_Float16)xr[mt][r];
    __syncthreads();

    // (b) hidden layers: drift = tanh(x@dW1^T+db1), diff = relu(x@fW1^T+fb1)
    _Float16(*hout)[Hh] = mat ? hf : hd;
#pragma unroll
    for (int mt2 = 0; mt2 < 4; ++mt2) {
      v8f acc;
#pragma unroll
      for (int r = 0; r < 8; ++r) acc[r] = b1v;
      int mrow = mt2 * 16 + l16;
#pragma unroll
      for (int ki = 0; ki < 8; ++ki) {
        v16h a = load_a_frag(&xbuf[0][0], Cc, mrow, ki * 32, half);
        acc = wmma_f16(a, bw1[ki], acc);
      }
#pragma unroll
      for (int r = 0; r < 8; ++r) {
        float v = mat ? fmaxf(acc[r], 0.f) : tanhf(acc[r]);
        hout[mt2 * 16 + r + 8 * half][nh] = (_Float16)v;
      }
    }
    __syncthreads();

    // (c) output layers + SDE + Kalman blend
#pragma unroll
    for (int mt = 0; mt < 4; ++mt) {
      v8f accd, accf;
#pragma unroll
      for (int r = 0; r < 8; ++r) { accd[r] = db2v; accf[r] = fb2v; }
      int mrow = mt * 16 + l16;
#pragma unroll
      for (int ki = 0; ki < 4; ++ki) {
        v16h ad = load_a_frag(&hd[0][0], Hh, mrow, ki * 32, half);
        accd = wmma_f16(ad, bwd2[ki], accd);
        v16h af = load_a_frag(&hf[0][0], Hh, mrow, ki * 32, half);
        accf = wmma_f16(af, bwf2[ki], accf);
      }
      size_t fidx = ((((size_t)t * 4 + mt) * 16 + ct) * 32 + lane) * 8;
      v8f kv8 = *(const v8f*)(kf + fidx);
      v8f zv8 = *(const v8f*)(zf + fidx);
#pragma unroll
      for (int r = 0; r < 8; ++r) {
        int m = mt * 16 + r + 8 * half;
        float diffv = softp(accf[r]);
        float nv = nz[((size_t)(t - 1) * Bb + m) * Cc + c];
        float xs = xr[mt][r] + accd[r] + diffv * nv;   // DT = sqrt_dt = 1
        float xn = xs + kv8[r] * (zv8[r] - xs);
        xr[mt][r] = xn;
        refined[((size_t)m * Cc + c) * Tt + t] = xn;
      }
    }
    __syncthreads();
  }
}

// ---------------------------------------------------------------------------
// Host launcher
// ---------------------------------------------------------------------------
extern "C" void kernel_launch(void* const* d_in, const int* in_sizes, int n_in,
                              void* d_out, int out_size, void* d_ws, size_t ws_size,
                              hipStream_t stream) {
  const float* z    = (const float*)d_in[0];
  const float* nz   = (const float*)d_in[1];
  const float* Wih  = (const float*)d_in[2];
  const float* Whh  = (const float*)d_in[3];
  const float* bih  = (const float*)d_in[4];
  const float* bhh  = (const float*)d_in[5];
  const float* fcW  = (const float*)d_in[6];
  const float* fcb  = (const float*)d_in[7];
  const float* dW1  = (const float*)d_in[8];
  const float* db1  = (const float*)d_in[9];
  const float* dW2  = (const float*)d_in[10];
  const float* db2  = (const float*)d_in[11];
  const float* fW1  = (const float*)d_in[12];
  const float* fb1  = (const float*)d_in[13];
  const float* fW2  = (const float*)d_in[14];
  const float* fb2  = (const float*)d_in[15];
  const float* nW1  = (const float*)d_in[16];
  const float* nb1  = (const float*)d_in[17];
  const float* nW2  = (const float*)d_in[18];
  const float* nb2  = (const float*)d_in[19];

  // workspace layout (bytes)
  constexpr size_t O_WIH  = 0;
  constexpr size_t O_WHH  = O_WIH  + (size_t)Gg * Cc * 2;
  constexpr size_t O_FCW  = O_WHH  + (size_t)Gg * Hh * 2;
  constexpr size_t O_DW1  = O_FCW  + (size_t)Cc * Hh * 2;
  constexpr size_t O_DW2  = O_DW1  + (size_t)Hh * Cc * 2;
  constexpr size_t O_FW1  = O_DW2  + (size_t)Cc * Hh * 2;
  constexpr size_t O_FW2  = O_FW1  + (size_t)Hh * Cc * 2;
  constexpr size_t O_NW1  = O_FW2  + (size_t)Cc * Hh * 2;
  constexpr size_t O_NW2  = O_NW1  + (size_t)Cc * Cc * 2;
  constexpr size_t O_BALL = O_NW2  + (size_t)Cc * Cc * 2;
  constexpr size_t O_ZT   = O_BALL + (size_t)Gg * 4;
  constexpr size_t O_GXF  = O_ZT   + (size_t)Tt * Bb * Cc * 2;
  constexpr size_t O_HALL = O_GXF  + (size_t)Tt * Bb * Gg * 4;
  constexpr size_t O_RES  = O_HALL + (size_t)Tt * Bb * Hh * 2;
  constexpr size_t O_NH1  = O_RES  + (size_t)Tt * Bb * Cc * 2;
  constexpr size_t O_KF   = O_NH1  + (size_t)Tt * Bb * Cc * 2;
  constexpr size_t O_ZF   = O_KF   + (size_t)Tt * Bb * Cc * 4;

  char* ws = (char*)d_ws;
  _Float16* wih16 = (_Float16*)(ws + O_WIH);
  _Float16* whh16 = (_Float16*)(ws + O_WHH);
  _Float16* fcw16 = (_Float16*)(ws + O_FCW);
  _Float16* dw116 = (_Float16*)(ws + O_DW1);
  _Float16* dw216 = (_Float16*)(ws + O_DW2);
  _Float16* fw116 = (_Float16*)(ws + O_FW1);
  _Float16* fw216 = (_Float16*)(ws + O_FW2);
  _Float16* nw116 = (_Float16*)(ws + O_NW1);
  _Float16* nw216 = (_Float16*)(ws + O_NW2);
  float*    ball  = (float*)   (ws + O_BALL);
  _Float16* zt16  = (_Float16*)(ws + O_ZT);
  float*    gxf   = (float*)   (ws + O_GXF);
  _Float16* hall  = (_Float16*)(ws + O_HALL);
  _Float16* res16 = (_Float16*)(ws + O_RES);
  _Float16* nh116 = (_Float16*)(ws + O_NH1);
  float*    kf    = (float*)   (ws + O_KF);
  float*    zf    = (float*)   (ws + O_ZF);

  float* refined = (float*)d_out;
  float* uncert  = (float*)d_out + (size_t)Bb * Cc * Tt;

  auto cvt = [&](const float* s, _Float16* d, int n) {
    k_cvt<<<(n + 255) / 256, 256, 0, stream>>>(s, d, n);
  };
  cvt(Wih, wih16, Gg * Cc);
  cvt(Whh, whh16, Gg * Hh);
  cvt(fcW, fcw16, Cc * Hh);
  cvt(dW1, dw116, Hh * Cc);
  cvt(dW2, dw216, Cc * Hh);
  cvt(fW1, fw116, Hh * Cc);
  cvt(fW2, fw216, Cc * Hh);
  cvt(nW1, nw116, Cc * Cc);
  cvt(nW2, nw216, Cc * Cc);
  k_badd<<<2, 256, 0, stream>>>(bih, bhh, ball);
  k_zt<<<(Tt * Bb * Cc + 255) / 256, 256, 0, stream>>>(z, zt16);
  k_zf<<<(Tt * Bb * Cc + 255) / 256, 256, 0, stream>>>(z, zf);

  k_gatesx<<<2048, 256, 0, stream>>>(zt16, wih16, ball, gxf);
  k_lstm<<<1, 512, 0, stream>>>(gxf, whh16, hall);
  k_fc_resid<<<1024, 256, 0, stream>>>(hall, fcw16, fcb, z, res16);
  k_noise1<<<1024, 256, 0, stream>>>(res16, nw116, nb1, nh116);
  k_noise2<<<1024, 256, 0, stream>>>(nh116, nw216, nb2, kf, uncert);
  k_filter<<<1, 512, 0, stream>>>(zf, nz, kf, dw116, db1, dw216, db2,
                                  fw116, fb1, fw216, fb2, refined, uncert);
  (void)in_sizes; (void)n_in; (void)out_size; (void)ws_size;
}
